// LlamaEncoderBackbone_17695265259625
// MI455X (gfx1250) — matrix operand order repaired and checked by reading, not compiled, not run on hardware
//
#include <hip/hip_runtime.h>
#include <hip/hip_bf16.h>

// ---------------------------------------------------------------------------
// Llama-2-ish NF4 backbone for MI455X (gfx1250), wave32 + WMMA f16.
// GEMMs: NF4 dequant -> f16 tile in LDS (VALU), activations staged with
// GLOBAL_LOAD_ASYNC_TO_LDS_B128 (ASYNCcnt), v_wmma_f32_16x16x32_f16 compute.
// Attention: WMMA flash-attention with online softmax, async K staging.
// ---------------------------------------------------------------------------

typedef __attribute__((ext_vector_type(16))) _Float16 v16h;
typedef __attribute__((ext_vector_type(8)))  _Float16 v8h;
typedef __attribute__((ext_vector_type(8)))  float    v8f;

#define LAYERS 2
#define DMODEL 4096
#define KVDIM  1024
#define FFDIM  14336
#define NHEAD  32
#define NKV    8
#define HDIM   128
#define BATCH  2
#define SEQ    1024
#define MROWS  (BATCH * SEQ)   // 2048

__constant__ float NF4_TAB[16] = {
    -1.0f, -0.6961928009986877f, -0.5250730514526367f, -0.39491748809814453f,
    -0.28444138169288635f, -0.18477343022823334f, -0.09105003625154495f, 0.0f,
    0.07958029955625534f, 0.16093020141124725f, 0.24611230194568634f,
    0.33791524171829224f, 0.44070982933044434f, 0.5626170039176941f,
    0.7229568362236023f, 1.0f};

// --- gfx1250 async global->LDS copy (ASYNCcnt) ------------------------------
__device__ __forceinline__ unsigned int lds_addr_of(const void* p) {
  return (unsigned int)(uintptr_t)(__attribute__((address_space(3))) const void*)p;
}
__device__ __forceinline__ void async_copy_b128(unsigned int lds_off,
                                                const void* gaddr) {
  asm volatile("global_load_async_to_lds_b128 %0, %1, off"
               :
               : "v"(lds_off), "v"((unsigned long long)(uintptr_t)gaddr)
               : "memory");
}
__device__ __forceinline__ void wait_async0() {
  asm volatile("s_wait_asynccnt 0" ::: "memory");
}

// --- WMMA fragment loads ----------------------------------------------------
// A fragment (16x32 f16): lane L -> row m = L&15; lane-half selects K subset:
// VGPR 0..3 hold K = sel*8 .. sel*8+7, VGPR 4..7 hold K = 16+sel*8 .. +7.
__device__ __forceinline__ v16h frag_a(const _Float16* base, int stride,
                                       int kbase, int lane) {
  const int m = lane & 15, sel = lane >> 4;
  const _Float16* p = base + m * stride + kbase + sel * 8;
  v8h lo = *(const v8h*)(p);
  v8h hi = *(const v8h*)(p + 16);
  v16h r;
#pragma unroll
  for (int i = 0; i < 8; ++i) { r[i] = lo[i]; r[i + 8] = hi[i]; }
  return r;
}

// B fragment (32x16 f16), storage is [N][K] (K contiguous): lane L -> col
// n = L&15, lane-half selects K = sel*8..+7 and 16+sel*8..+7 (same as A).
__device__ __forceinline__ v16h frag_b(const _Float16* base, int stride,
                                       int kbase, int lane) {
  const int n = lane & 15, kb = (lane >> 4) * 8;
  const _Float16* p = base + n * stride + kbase + kb;
  v8h lo = *(const v8h*)(p);
  v8h hi = *(const v8h*)(p + 16);
  v16h r;
#pragma unroll
  for (int i = 0; i < 8; ++i) { r[i] = lo[i]; r[i + 8] = hi[i]; }
  return r;
}

// --- Embedding gather -------------------------------------------------------
__global__ __launch_bounds__(256) void embed_kernel(
    const int* __restrict__ ids, const float* __restrict__ emb,
    float* __restrict__ h) {
  const int row = blockIdx.x;
  const int id = ids[row];
  const float* src = emb + (long)id * DMODEL;
  float* dst = h + (long)row * DMODEL;
  for (int j = threadIdx.x; j < DMODEL; j += 256) dst[j] = src[j];
}

// --- RMSNorm (f32 in, f16 or f32 out) --------------------------------------
template <bool H16>
__global__ __launch_bounds__(256) void rmsnorm_kernel(
    const float* __restrict__ h, const float* __restrict__ w,
    _Float16* __restrict__ outH, float* __restrict__ outF) {
  __shared__ float red[8];
  const int row = blockIdx.x, tid = threadIdx.x;
  const float* hp = h + (long)row * DMODEL;
  float vals[16];
  float ss = 0.f;
#pragma unroll
  for (int j = 0; j < 16; ++j) {
    vals[j] = hp[tid + j * 256];
    ss += vals[j] * vals[j];
  }
#pragma unroll
  for (int m = 1; m < 32; m <<= 1) ss += __shfl_xor(ss, m, 32);
  if ((tid & 31) == 0) red[tid >> 5] = ss;
  __syncthreads();
  float tot = 0.f;
#pragma unroll
  for (int i = 0; i < 8; ++i) tot += red[i];
  const float sc = rsqrtf(tot / (float)DMODEL + 1e-5f);
#pragma unroll
  for (int j = 0; j < 16; ++j) {
    const float o = vals[j] * sc * w[tid + j * 256];
    if (H16)
      outH[(long)row * DMODEL + tid + j * 256] = (_Float16)o;
    else
      outF[(long)row * DMODEL + tid + j * 256] = o;
  }
}

// --- Fused NF4-dequant GEMM:  out[M,N] = X[M,K] * W[N,K]^T ------------------
// Block tile 64(M) x 128(N), K-chunk 64. 8 waves: wave = (wm 0..3, wn 0..1),
// each wave accumulates 16x64 (4 x 16x16 frags) so one A-frag feeds 4 WMMAs.
// EPI 0: f32 out    1: f16 out    2: f32 out = aux + acc (residual)
// EPI 3: f16 out = silu(aux) * acc   (fused SwiGLU for up-proj)
template <int EPI>
__global__ __launch_bounds__(256) void gemm_nf4(
    const _Float16* __restrict__ X, const unsigned char* __restrict__ Widx,
    const float* __restrict__ Wamax, float* __restrict__ outF,
    _Float16* __restrict__ outH, const float* __restrict__ aux, int M, int N,
    int K) {
  __shared__ _Float16 Xs[64 * 72];
  __shared__ _Float16 Ws[128 * 72];
  const int tid = threadIdx.x;
  const int lane = tid & 31;
  const int wave = tid >> 5;
  const int wm = wave & 3;   // 16-row group
  const int wn = wave >> 2;  // 64-col half
  const int m0 = blockIdx.y * 64;
  const int n0 = blockIdx.x * 128;
  const int Kb = K >> 6;

  v8f acc[4] = {};

  const int xr = tid >> 2;   // 0..63  X tile row
  const int xs = tid & 3;    // 16-half segment
  const int wr = tid >> 1;   // 0..127 W tile row
  const int wsg = tid & 1;   // 32-byte segment

  for (int k0 = 0; k0 < K; k0 += 64) {
    // Stage X tile via async global->LDS (2 x b128 per thread)
    {
      const _Float16* src = X + (long)(m0 + xr) * K + k0 + xs * 16;
      const unsigned int la = lds_addr_of(Xs + xr * 72 + xs * 16);
      async_copy_b128(la, src);
      async_copy_b128(la + 16, src + 8);
    }
    // Dequant W tile NF4 -> f16, stored [N][K] (32 elements per thread)
    {
      const unsigned char* wsrc = Widx + (long)(n0 + wr) * K + k0 + wsg * 32;
      const float am = Wamax[(long)(n0 + wr) * Kb + (k0 >> 6)];
      _Float16* dst = Ws + wr * 72 + wsg * 32;
#pragma unroll
      for (int q = 0; q < 2; ++q) {
        uint4 raw = *(const uint4*)(wsrc + q * 16);
        const unsigned char* bytes = (const unsigned char*)&raw;
#pragma unroll
        for (int i = 0; i < 16; ++i)
          dst[q * 16 + i] = (_Float16)(NF4_TAB[bytes[i] & 15] * am);
      }
      if (k0 + 64 < K) __builtin_prefetch(wsrc + 64, 0, 1);  // global_prefetch_b8
    }
    wait_async0();
    __syncthreads();
#pragma unroll
    for (int kk = 0; kk < 2; ++kk) {
      v16h a = frag_a(Xs + wm * 16 * 72, 72, kk * 32, lane);
#pragma unroll
      for (int nf = 0; nf < 4; ++nf) {
        v16h b = frag_b(Ws + (wn * 64 + nf * 16) * 72, 72, kk * 32, lane);
        acc[nf] = __builtin_amdgcn_wmma_f32_16x16x32_f16(
            false, a, false, b, (short)0, acc[nf], false, false);
      }
    }
    __syncthreads();
  }

  const int hi = lane >> 4, nc = lane & 15;
#pragma unroll
  for (int nf = 0; nf < 4; ++nf) {
#pragma unroll
    for (int vg = 0; vg < 8; ++vg) {
      const int row = m0 + wm * 16 + vg + 8 * hi;
      const int col = n0 + wn * 64 + nf * 16 + nc;
      const long idx = (long)row * N + col;
      const float v = acc[nf][vg];
      if (EPI == 0) {
        outF[idx] = v;
      } else if (EPI == 1) {
        outH[idx] = (_Float16)v;
      } else if (EPI == 2) {
        outF[idx] = aux[idx] + v;
      } else {
        const float g = aux[idx];
        const float sg = g / (1.f + __expf(-g));
        outH[idx] = (_Float16)(sg * v);
      }
    }
  }
}

// --- RoPE (in place on f16 activations, layout (B*S, nheads, HDIM)) --------
__global__ __launch_bounds__(64) void rope_kernel(_Float16* __restrict__ x,
                                                  int nheads) {
  const int row = blockIdx.x;   // b*S + s
  const int head = blockIdx.y;
  const int i = threadIdx.x;    // 0..63 (HDIM/2)
  const float pos = (float)(row & (SEQ - 1));
  const float freq = __powf(500000.0f, -(float)(2 * i) / (float)HDIM);
  const float a = pos * freq;
  const float c = __cosf(a), s = __sinf(a);
  _Float16* p = x + ((long)row * nheads + head) * HDIM;
  const float x1 = (float)p[i], x2 = (float)p[i + 64];
  p[i] = (_Float16)(x1 * c - x2 * s);
  p[i + 64] = (_Float16)(x2 * c + x1 * s);
}

// --- Flash attention (causal, GQA 4:1), WMMA for QK^T and PV ----------------
__global__ __launch_bounds__(256) void attention_kernel(
    const _Float16* __restrict__ q, const _Float16* __restrict__ k,
    const _Float16* __restrict__ v, _Float16* __restrict__ ctx) {
  __shared__ _Float16 Ks[64 * 136];   // [key][hd]
  __shared__ _Float16 Vt[128 * 72];   // [hd][key]  (transposed)
  __shared__ _Float16 Ps[64 * 72];    // softmax probs [row][key]
  __shared__ float row_scale[64];
  __shared__ float row_l[64];

  const int tid = threadIdx.x, lane = tid & 31, wave = tid >> 5;
  const int wq = wave & 3;       // 16-row group within q tile
  const int half = wave >> 2;    // HD half for PV (0 or 1)
  const int qt = blockIdx.x;
  const int h = blockIdx.y;
  const int b = blockIdx.z;
  const int kvh = h >> 2;        // H/HK = 4
  const int qbase = qt * 64;
  const int hi = lane >> 4, nc = lane & 15;

  // Preload Q fragments (score waves): 16 rows x 128 (4 x 16x32 A-frags)
  v16h qf[4] = {};
  if (half == 0) {
    const _Float16* qrow =
        q + (((long)b * SEQ + qbase + wq * 16) * NHEAD + h) * HDIM;
#pragma unroll
    for (int kk = 0; kk < 4; ++kk)
      qf[kk] = frag_a(qrow, NHEAD * HDIM, kk * 32, lane);
  }

  v8f acc[4] = {};
  float mrow[8], lrow[8];
#pragma unroll
  for (int i = 0; i < 8; ++i) { mrow[i] = -3.0e38f; lrow[i] = 0.f; }

  for (int kc = 0; kc <= qt; ++kc) {
    // Stage K chunk (async global->LDS) and transposed V chunk
    {
      const int r = tid >> 2, sg = tid & 3;
      const _Float16* ksrc =
          k + (((long)b * SEQ + kc * 64 + r) * NKV + kvh) * HDIM + sg * 32;
      const unsigned int la = lds_addr_of(Ks + r * 136 + sg * 32);
#pragma unroll
      for (int j = 0; j < 4; ++j) async_copy_b128(la + j * 16, ksrc + j * 8);
      const _Float16* vsrc =
          v + (((long)b * SEQ + kc * 64 + r) * NKV + kvh) * HDIM + sg * 32;
#pragma unroll
      for (int j = 0; j < 32; ++j) Vt[(sg * 32 + j) * 72 + r] = vsrc[j];
    }
    wait_async0();
    __syncthreads();

    if (half == 0) {
      // scores S = Q . K^T  (4 n-frags x 4 k-steps)
      v8f sc[4] = {};
#pragma unroll
      for (int nf = 0; nf < 4; ++nf) {
#pragma unroll
        for (int kk = 0; kk < 4; ++kk) {
          v16h bf = frag_b(Ks + nf * 16 * 136, 136, kk * 32, lane);
          sc[nf] = __builtin_amdgcn_wmma_f32_16x16x32_f16(
              false, qf[kk], false, bf, (short)0, sc[nf], false, false);
        }
      }
      // scale + causal mask
#pragma unroll
      for (int nf = 0; nf < 4; ++nf) {
        const int colg = kc * 64 + nf * 16 + nc;
#pragma unroll
        for (int vg = 0; vg < 8; ++vg) {
          const int rowg = qbase + wq * 16 + vg + 8 * hi;
          const float s = sc[nf][vg] * 0.08838834764831845f;  // 1/sqrt(128)
          sc[nf][vg] = (colg <= rowg) ? s : -1.0e30f;
        }
      }
      // online softmax; row lives across one 16-lane half
#pragma unroll
      for (int vg = 0; vg < 8; ++vg) {
        float mx = fmaxf(fmaxf(sc[0][vg], sc[1][vg]),
                         fmaxf(sc[2][vg], sc[3][vg]));
#pragma unroll
        for (int msk = 1; msk < 16; msk <<= 1)
          mx = fmaxf(mx, __shfl_xor(mx, msk, 32));
        const float mnew = fmaxf(mrow[vg], mx);
        const float resc = __expf(mrow[vg] - mnew);
        const int rl = wq * 16 + vg + 8 * hi;
        float sum = 0.f;
#pragma unroll
        for (int nf = 0; nf < 4; ++nf) {
          const float p = __expf(sc[nf][vg] - mnew);
          sum += p;
          Ps[rl * 72 + nf * 16 + nc] = (_Float16)p;
        }
#pragma unroll
        for (int msk = 1; msk < 16; msk <<= 1) sum += __shfl_xor(sum, msk, 32);
        lrow[vg] = lrow[vg] * resc + sum;
        mrow[vg] = mnew;
        if (nc == 0) row_scale[rl] = resc;
      }
    }
    __syncthreads();

    // Rescale accumulators, then O += P . V
#pragma unroll
    for (int nf = 0; nf < 4; ++nf)
#pragma unroll
      for (int vg = 0; vg < 8; ++vg)
        acc[nf][vg] *= row_scale[wq * 16 + vg + 8 * hi];
#pragma unroll
    for (int kk = 0; kk < 2; ++kk) {
      v16h pa = frag_a(Ps + wq * 16 * 72, 72, kk * 32, lane);
#pragma unroll
      for (int nf = 0; nf < 4; ++nf) {
        v16h bf = frag_b(Vt + (half * 64 + nf * 16) * 72, 72, kk * 32, lane);
        acc[nf] = __builtin_amdgcn_wmma_f32_16x16x32_f16(
            false, pa, false, bf, (short)0, acc[nf], false, false);
      }
    }
    __syncthreads();
  }

  if (half == 0) {
#pragma unroll
    for (int vg = 0; vg < 8; ++vg)
      if (nc == 0) row_l[wq * 16 + vg + 8 * hi] = lrow[vg];
  }
  __syncthreads();

#pragma unroll
  for (int nf = 0; nf < 4; ++nf)
#pragma unroll
    for (int vg = 0; vg < 8; ++vg) {
      const int rl = wq * 16 + vg + 8 * hi;
      const int srow = qbase + rl;
      const int hd = half * 64 + nf * 16 + nc;
      const float o = acc[nf][vg] / row_l[rl];
      ctx[(((long)b * SEQ + srow) * NHEAD + h) * HDIM + hd] = (_Float16)o;
    }
}

// --- Masked mean pooling ----------------------------------------------------
__global__ __launch_bounds__(256) void pool_kernel(
    const float* __restrict__ normed, const int* __restrict__ mask,
    float* __restrict__ out) {
  const int d = blockIdx.x * 256 + threadIdx.x;
  const int b = blockIdx.y;
  float s = 0.f, c = 0.f;
  for (int t = 0; t < SEQ; ++t) {
    const float mm = (mask[b * SEQ + t] > 0) ? 1.f : 0.f;
    s += mm * normed[((long)b * SEQ + t) * DMODEL + d];
    c += mm;
  }
  out[b * DMODEL + d] = s / fmaxf(c, 1e-9f);
}

// ---------------------------------------------------------------------------
extern "C" void kernel_launch(void* const* d_in, const int* in_sizes, int n_in,
                              void* d_out, int out_size, void* d_ws,
                              size_t ws_size, hipStream_t stream) {
  (void)in_sizes; (void)n_in; (void)out_size; (void)ws_size;
  const int* input_ids = (const int*)d_in[0];
  const int* attn_mask = (const int*)d_in[1];
  const float* embed = (const float*)d_in[2];
  const unsigned char* q_idx = (const unsigned char*)d_in[3];
  const float* q_am = (const float*)d_in[4];
  const unsigned char* k_idx = (const unsigned char*)d_in[5];
  const float* k_am = (const float*)d_in[6];
  const unsigned char* v_idx = (const unsigned char*)d_in[7];
  const float* v_am = (const float*)d_in[8];
  const unsigned char* o_idx = (const unsigned char*)d_in[9];
  const float* o_am = (const float*)d_in[10];
  const unsigned char* g_idx = (const unsigned char*)d_in[11];
  const float* g_am = (const float*)d_in[12];
  const unsigned char* u_idx = (const unsigned char*)d_in[13];
  const float* u_am = (const float*)d_in[14];
  const unsigned char* dn_idx = (const unsigned char*)d_in[15];
  const float* dn_am = (const float*)d_in[16];
  const float* ln1 = (const float*)d_in[17];
  const float* ln2 = (const float*)d_in[18];
  const float* lnf = (const float*)d_in[19];

  char* ws = (char*)d_ws;
  size_t off = 0;
  auto alloc = [&](size_t bytes) {
    void* p = ws + off;
    off = (off + bytes + 255) & ~(size_t)255;
    return p;
  };
  float* h = (float*)alloc((size_t)MROWS * DMODEL * 4);
  _Float16* x16 = (_Float16*)alloc((size_t)MROWS * DMODEL * 2);
  _Float16* q16 = (_Float16*)alloc((size_t)MROWS * DMODEL * 2);
  _Float16* k16 = (_Float16*)alloc((size_t)MROWS * KVDIM * 2);
  _Float16* v16 = (_Float16*)alloc((size_t)MROWS * KVDIM * 2);
  _Float16* c16 = (_Float16*)alloc((size_t)MROWS * DMODEL * 2);
  float* g32 = (float*)alloc((size_t)MROWS * FFDIM * 4);
  _Float16* a16 = (_Float16*)alloc((size_t)MROWS * FFDIM * 2);

  embed_kernel<<<MROWS, 256, 0, stream>>>(input_ids, embed, h);

  for (int l = 0; l < LAYERS; ++l) {
    const unsigned char* qi = q_idx + (size_t)l * DMODEL * DMODEL;
    const float* qa = q_am + (size_t)l * DMODEL * DMODEL / 64;
    const unsigned char* ki = k_idx + (size_t)l * KVDIM * DMODEL;
    const float* ka = k_am + (size_t)l * KVDIM * DMODEL / 64;
    const unsigned char* vi = v_idx + (size_t)l * KVDIM * DMODEL;
    const float* va = v_am + (size_t)l * KVDIM * DMODEL / 64;
    const unsigned char* oi = o_idx + (size_t)l * DMODEL * DMODEL;
    const float* oa = o_am + (size_t)l * DMODEL * DMODEL / 64;
    const unsigned char* gi = g_idx + (size_t)l * FFDIM * DMODEL;
    const float* ga = g_am + (size_t)l * FFDIM * DMODEL / 64;
    const unsigned char* ui = u_idx + (size_t)l * FFDIM * DMODEL;
    const float* ua = u_am + (size_t)l * FFDIM * DMODEL / 64;
    const unsigned char* di = dn_idx + (size_t)l * DMODEL * FFDIM;
    const float* da = dn_am + (size_t)l * DMODEL * FFDIM / 64;

    rmsnorm_kernel<true><<<MROWS, 256, 0, stream>>>(h, ln1 + l * DMODEL, x16,
                                                    nullptr);
    gemm_nf4<1><<<dim3(DMODEL / 128, MROWS / 64), 256, 0, stream>>>(
        x16, qi, qa, nullptr, q16, nullptr, MROWS, DMODEL, DMODEL);
    gemm_nf4<1><<<dim3(KVDIM / 128, MROWS / 64), 256, 0, stream>>>(
        x16, ki, ka, nullptr, k16, nullptr, MROWS, KVDIM, DMODEL);
    gemm_nf4<1><<<dim3(KVDIM / 128, MROWS / 64), 256, 0, stream>>>(
        x16, vi, va, nullptr, v16, nullptr, MROWS, KVDIM, DMODEL);
    rope_kernel<<<dim3(MROWS, NHEAD), 64, 0, stream>>>(q16, NHEAD);
    rope_kernel<<<dim3(MROWS, NKV), 64, 0, stream>>>(k16, NKV);
    attention_kernel<<<dim3(SEQ / 64, NHEAD, BATCH), 256, 0, stream>>>(
        q16, k16, v16, c16);
    gemm_nf4<2><<<dim3(DMODEL / 128, MROWS / 64), 256, 0, stream>>>(
        c16, oi, oa, h, nullptr, h, MROWS, DMODEL, DMODEL);
    rmsnorm_kernel<true><<<MROWS, 256, 0, stream>>>(h, ln2 + l * DMODEL, x16,
                                                    nullptr);
    gemm_nf4<0><<<dim3(FFDIM / 128, MROWS / 64), 256, 0, stream>>>(
        x16, gi, ga, g32, nullptr, nullptr, MROWS, FFDIM, DMODEL);
    gemm_nf4<3><<<dim3(FFDIM / 128, MROWS / 64), 256, 0, stream>>>(
        x16, ui, ua, nullptr, a16, g32, MROWS, FFDIM, DMODEL);
    gemm_nf4<2><<<dim3(DMODEL / 128, MROWS / 64), 256, 0, stream>>>(
        a16, di, da, h, nullptr, h, MROWS, DMODEL, FFDIM);
  }

  // final rmsnorm (f32, reuse g32 as scratch) + masked mean pool
  rmsnorm_kernel<false><<<MROWS, 256, 0, stream>>>(h, lnf, nullptr, g32);
  pool_kernel<<<dim3(DMODEL / 256, BATCH), 256, 0, stream>>>(g32, attn_mask,
                                                             (float*)d_out);
}
